// EncoderDecoder_5050881540127
// MI455X (gfx1250) — matrix operand order repaired and checked by reading, not compiled, not run on hardware
//
#include <hip/hip_runtime.h>
#include <hip/hip_bf16.h>

// ---------------- types ----------------
typedef __attribute__((ext_vector_type(16))) __bf16 v16bf;
typedef __attribute__((ext_vector_type(8)))  __bf16 v8bf;
typedef __attribute__((ext_vector_type(8)))  float  v8f;

#define Bsz 256
#define Hs  512
#define Ls  32
#define Vs  10000
#define H2  1024
#define H3  1536

// round-to-nearest-even f32 -> bf16 without relying on bf16 VALU conversions
__device__ __forceinline__ __bf16 f2bf(float f) {
    unsigned u = __builtin_bit_cast(unsigned, f);
    unsigned r = (u + 0x7FFFu + ((u >> 16) & 1u)) >> 16;
    unsigned short s = (unsigned short)r;
    return __builtin_bit_cast(__bf16, s);
}

__device__ __forceinline__ float sigf(float x) { return 1.0f / (1.0f + __expf(-x)); }

// ---------------- f32 -> bf16 cast ----------------
__global__ void cast_bf16_kernel(const float* __restrict__ src, __bf16* __restrict__ dst, int n) {
    int i = blockIdx.x * blockDim.x + threadIdx.x;
    if (i < n) dst[i] = f2bf(src[i]);
}

// ---------------- generic WMMA GEMM: C[M,N] = A[M,K] * B[N,K]^T + bias ----------------
// A row-major bf16, B row-major bf16 (weight layout [N,K]).
// 4 waves/block; each wave owns one N-tile and MT M-tiles (B fragment reused MT times).
// A strip (MT*16 x K) staged into LDS with CDNA5 async global->LDS copies.
template<int MT>
__global__ void __launch_bounds__(128)
gemm_bf16_wmma(const __bf16* __restrict__ A, const __bf16* __restrict__ Bm,
               const float* __restrict__ bias,
               float* __restrict__ Cf, __bf16* __restrict__ Cbf,
               int M, int N, int K, int relu) {
    extern __shared__ __bf16 ldsA[]; // (MT*16) x K bf16  (64 KB for both shapes used)
    const int tid   = threadIdx.x;
    const int wave  = tid >> 5;
    const int lane  = tid & 31;
    const int m0    = blockIdx.y * (MT * 16);
    const int ntile = (blockIdx.x << 2) + wave;

    // ---- async-stage the (MT*16) x K A-strip into LDS (16B per lane per op) ----
    const int tot = (MT * 16) * K;            // uniform trip count across all lanes
    for (int e = tid * 8; e < tot; e += 128 * 8) {
        int r = e / K, c = e - r * K;
        unsigned lds_addr = (unsigned)(unsigned long long)(ldsA + r * K + c); // LDS byte offset (addr[31:0])
        unsigned long long gaddr = (unsigned long long)(A + (size_t)(m0 + r) * K + c);
        asm volatile("global_load_async_to_lds_b128 %0, %1, off"
                     :: "v"(lds_addr), "v"(gaddr) : "memory");
    }
    asm volatile("s_wait_asynccnt 0" ::: "memory");
    __syncthreads();

    if ((ntile << 4) < N) {
        const int l15  = lane & 15;     // A row-in-tile / B col / D col
        const int half = lane >> 4;     // K-half selector
        v8f acc[MT];
        #pragma unroll
        for (int i = 0; i < MT; ++i) acc[i] = (v8f){};
        const __bf16* brow = Bm + (size_t)((ntile << 4) + l15) * K + half * 16;
        const int arow_off = l15 * K + half * 16;
        for (int kt = 0; kt < K; kt += 32) {
            __builtin_prefetch(brow + kt + 128, 0, 1);      // global_prefetch_b8
            v16bf bf = *(const v16bf*)(brow + kt);          // K = h*16 + 0..15 contiguous
            #pragma unroll
            for (int mt = 0; mt < MT; ++mt) {
                const __bf16* ap = ldsA + (size_t)(mt * 16) * K + arow_off + kt;
                v8bf  alo = *(const v8bf*)(ap);             // K = h*16 + 0..7
                v8bf  ahi = *(const v8bf*)(ap + 8);         // K = h*16 + 8..15
                v16bf af  = __builtin_shufflevector(alo, ahi,
                              0,1,2,3,4,5,6,7,8,9,10,11,12,13,14,15);
                acc[mt] = __builtin_amdgcn_wmma_f32_16x16x32_bf16(false, af, false, bf,
                                                                  (short)0, acc[mt], false, false);
            }
        }
        const int col = (ntile << 4) + l15;
        const float bv = bias ? bias[col] : 0.0f;
        #pragma unroll
        for (int mt = 0; mt < MT; ++mt) {
            #pragma unroll
            for (int r = 0; r < 8; ++r) {
                int mrow = m0 + mt * 16 + r + (half << 3);
                float v = acc[mt][r] + bv;
                if (relu) v = fmaxf(v, 0.0f);
                if (Cf)  Cf [(size_t)mrow * N + col] = v;
                if (Cbf) Cbf[(size_t)mrow * N + col] = f2bf(v);
            }
        }
    }
}

// ---------------- embedding gather -> bf16 [B,H] ----------------
__global__ void gather_embed_kernel(const int* __restrict__ toks, const float* __restrict__ embed,
                                    __bf16* __restrict__ out, int t, int dec_mode) {
    int idx = blockIdx.x * blockDim.x + threadIdx.x; // B*H
    int b = idx >> 9, h = idx & (Hs - 1);
    int tok = dec_mode ? ((t == 0) ? 1 : toks[b * Ls + t - 1]) : toks[b * Ls + t];
    out[idx] = f2bf(embed[(size_t)tok * Hs + h]);
}

// ---------------- fused GRU cell ----------------
__global__ void gru_fuse_kernel(const float* __restrict__ gi, const float* __restrict__ gh,
                                float* __restrict__ h, __bf16* __restrict__ h_bf,
                                float* __restrict__ enc_out_slice) {
    int idx = blockIdx.x * blockDim.x + threadIdx.x; // B*H
    int b = idx >> 9, j = idx & (Hs - 1);
    const float* gib = gi + (size_t)b * H3;
    const float* ghb = gh + (size_t)b * H3;
    float r = sigf(gib[j]          + ghb[j]);
    float z = sigf(gib[Hs + j]     + ghb[Hs + j]);
    float n = tanhf(gib[2*Hs + j] + r * ghb[2*Hs + j]);
    float hv = (1.0f - z) * n + z * h[idx];
    h[idx] = hv;
    h_bf[idx] = f2bf(hv);
    if (enc_out_slice) enc_out_slice[idx] = hv;
}

// ---------------- concat [emb_bf | f2bf(h)] -> [B, 2H] bf16 ----------------
__global__ void cat_eh_kernel(const __bf16* __restrict__ emb_bf, const float* __restrict__ hf,
                              __bf16* __restrict__ cat) {
    int idx = blockIdx.x * blockDim.x + threadIdx.x; // B*H
    int b = idx >> 9, h = idx & (Hs - 1);
    cat[(size_t)b * H2 + h]      = emb_bf[idx];
    cat[(size_t)b * H2 + Hs + h] = f2bf(hf[idx]);
}

// ---------------- wave-wide softmax over L=32 (one lane per column) ----------------
__global__ void softmax32_kernel(float* __restrict__ aw) {
    int row = blockIdx.x, lane = threadIdx.x;
    float v = aw[row * Ls + lane];
    float m = v;
    #pragma unroll
    for (int off = 16; off > 0; off >>= 1) m = fmaxf(m, __shfl_xor(m, off, 32));
    float e = __expf(v - m), s = e;
    #pragma unroll
    for (int off = 16; off > 0; off >>= 1) s += __shfl_xor(s, off, 32);
    aw[row * Ls + lane] = e / s;
}

// ---------------- applied = einsum('bl,lbh->bh'); fused concat [emb | applied] ----------------
__global__ void attn_apply_cat_kernel(const float* __restrict__ aw, const float* __restrict__ enc_outs,
                                      const __bf16* __restrict__ emb_bf, __bf16* __restrict__ cat) {
    int idx = blockIdx.x * blockDim.x + threadIdx.x; // B*H
    int b = idx >> 9, h = idx & (Hs - 1);
    float s = 0.0f;
    #pragma unroll 8
    for (int l = 0; l < Ls; ++l)
        s += aw[b * Ls + l] * enc_outs[((size_t)l * Bsz + b) * Hs + h];
    cat[(size_t)b * H2 + h]      = emb_bf[idx];
    cat[(size_t)b * H2 + Hs + h] = f2bf(s);
}

// ---------------- row-wise log_softmax over V ----------------
__global__ void log_softmax_kernel(const float* __restrict__ logits, float* __restrict__ out, int Vn) {
    __shared__ float red[256];
    int row = blockIdx.x, tid = threadIdx.x;
    const float* p = logits + (size_t)row * Vn;
    float mx = -3.4e38f;
    for (int i = tid; i < Vn; i += 256) mx = fmaxf(mx, p[i]);
    red[tid] = mx; __syncthreads();
    for (int s = 128; s > 0; s >>= 1) { if (tid < s) red[tid] = fmaxf(red[tid], red[tid + s]); __syncthreads(); }
    mx = red[0]; __syncthreads();
    float sum = 0.0f;
    for (int i = tid; i < Vn; i += 256) sum += __expf(p[i] - mx);
    red[tid] = sum; __syncthreads();
    for (int s = 128; s > 0; s >>= 1) { if (tid < s) red[tid] += red[tid + s]; __syncthreads(); }
    float lse = __logf(red[0]) + mx;
    float* o = out + (size_t)row * Vn;
    for (int i = tid; i < Vn; i += 256) o[i] = p[i] - lse;
}

// ---------------- init hidden ----------------
__global__ void zero_h_kernel(float* __restrict__ h, __bf16* __restrict__ hb) {
    int idx = blockIdx.x * blockDim.x + threadIdx.x;
    h[idx] = 0.0f; hb[idx] = f2bf(0.0f);
}

// ================= host =================
static void launch_gemm(const __bf16* A, const __bf16* Bm, const float* bias,
                        float* Cf, __bf16* Cbf, int M, int N, int K, int relu, hipStream_t s) {
    const int ntiles = N / 16;
    if (K == 512) {                                    // MT=4 -> 64 KB LDS A-strip
        dim3 grid((ntiles + 3) / 4, M / 64);
        size_t shmem = (size_t)64 * K * sizeof(__bf16);
        gemm_bf16_wmma<4><<<grid, 128, shmem, s>>>(A, Bm, bias, Cf, Cbf, M, N, K, relu);
    } else {                                           // K==1024: MT=2 -> 64 KB LDS A-strip
        dim3 grid((ntiles + 3) / 4, M / 32);
        size_t shmem = (size_t)32 * K * sizeof(__bf16);
        gemm_bf16_wmma<2><<<grid, 128, shmem, s>>>(A, Bm, bias, Cf, Cbf, M, N, K, relu);
    }
}

extern "C" void kernel_launch(void* const* d_in, const int* in_sizes, int n_in,
                              void* d_out, int out_size, void* d_ws, size_t ws_size,
                              hipStream_t stream) {
    const int*   input_tensor  = (const int*)d_in[0];
    const int*   target_tensor = (const int*)d_in[1];
    const float* enc_embed = (const float*)d_in[2];
    const float* enc_w_ih  = (const float*)d_in[3];
    const float* enc_w_hh  = (const float*)d_in[4];
    const float* enc_b_ih  = (const float*)d_in[5];
    const float* enc_b_hh  = (const float*)d_in[6];
    const float* dec_embed = (const float*)d_in[7];
    const float* attn_w    = (const float*)d_in[8];
    const float* attn_b    = (const float*)d_in[9];
    const float* comb_w    = (const float*)d_in[10];
    const float* comb_b    = (const float*)d_in[11];
    const float* dec_w_ih  = (const float*)d_in[12];
    const float* dec_w_hh  = (const float*)d_in[13];
    const float* dec_b_ih  = (const float*)d_in[14];
    const float* dec_b_hh  = (const float*)d_in[15];
    const float* out_w     = (const float*)d_in[16];
    const float* out_b     = (const float*)d_in[17];
    float* out = (float*)d_out;

    // bump allocator over d_ws
    char* ws = (char*)d_ws;
    size_t off = 0;
    auto alloc = [&](size_t bytes) -> void* {
        off = (off + 255) & ~(size_t)255;
        void* p = ws + off;
        off += bytes;
        return p;
    };

    __bf16* enc_w_ih_bf = (__bf16*)alloc((size_t)H3 * Hs * 2);
    __bf16* enc_w_hh_bf = (__bf16*)alloc((size_t)H3 * Hs * 2);
    __bf16* dec_w_ih_bf = (__bf16*)alloc((size_t)H3 * Hs * 2);
    __bf16* dec_w_hh_bf = (__bf16*)alloc((size_t)H3 * Hs * 2);
    __bf16* comb_w_bf   = (__bf16*)alloc((size_t)Hs * H2 * 2);
    __bf16* attn_w_bf   = (__bf16*)alloc((size_t)Ls * H2 * 2);
    __bf16* out_w_bf    = (__bf16*)alloc((size_t)Vs * Hs * 2);
    float*  enc_outs    = (float*) alloc((size_t)Ls * Bsz * Hs * 4);
    float*  h_f32       = (float*) alloc((size_t)Bsz * Hs * 4);
    __bf16* h_bf        = (__bf16*)alloc((size_t)Bsz * Hs * 2);
    __bf16* x_bf        = (__bf16*)alloc((size_t)Bsz * Hs * 2);
    __bf16* emb_bf      = (__bf16*)alloc((size_t)Bsz * Hs * 2);
    float*  gi          = (float*) alloc((size_t)Bsz * H3 * 4);
    float*  gh          = (float*) alloc((size_t)Bsz * H3 * 4);
    __bf16* cat_eh      = (__bf16*)alloc((size_t)Bsz * H2 * 2);
    __bf16* cat_comb    = (__bf16*)alloc((size_t)Bsz * H2 * 2);
    float*  aw          = (float*) alloc((size_t)Bsz * Ls * 4);
    float*  logits      = (float*) alloc((size_t)Bsz * Vs * 4);
    (void)ws_size; (void)in_sizes; (void)n_in; (void)out_size;

    // per-call weight casts (no caching allowed)
    struct { const float* s; __bf16* d; int n; } casts[] = {
        { enc_w_ih, enc_w_ih_bf, H3 * Hs }, { enc_w_hh, enc_w_hh_bf, H3 * Hs },
        { dec_w_ih, dec_w_ih_bf, H3 * Hs }, { dec_w_hh, dec_w_hh_bf, H3 * Hs },
        { comb_w,   comb_w_bf,   Hs * H2 }, { attn_w,   attn_w_bf,   Ls * H2 },
        { out_w,    out_w_bf,    Vs * Hs },
    };
    for (auto& c : casts)
        cast_bf16_kernel<<<(c.n + 255) / 256, 256, 0, stream>>>(c.s, c.d, c.n);

    const int BH_BLOCKS = (Bsz * Hs) / 256; // 512
    zero_h_kernel<<<BH_BLOCKS, 256, 0, stream>>>(h_f32, h_bf);

    // ---------------- encoder ----------------
    for (int t = 0; t < Ls; ++t) {
        gather_embed_kernel<<<BH_BLOCKS, 256, 0, stream>>>(input_tensor, enc_embed, x_bf, t, 0);
        launch_gemm(x_bf, enc_w_ih_bf, enc_b_ih, gi, nullptr, Bsz, H3, Hs, 0, stream);
        launch_gemm(h_bf, enc_w_hh_bf, enc_b_hh, gh, nullptr, Bsz, H3, Hs, 0, stream);
        gru_fuse_kernel<<<BH_BLOCKS, 256, 0, stream>>>(gi, gh, h_f32, h_bf,
                                                       enc_outs + (size_t)t * Bsz * Hs);
    }

    // ---------------- decoder (teacher forcing) ----------------
    for (int t = 0; t < Ls; ++t) {
        gather_embed_kernel<<<BH_BLOCKS, 256, 0, stream>>>(target_tensor, dec_embed, emb_bf, t, 1);
        cat_eh_kernel<<<BH_BLOCKS, 256, 0, stream>>>(emb_bf, h_f32, cat_eh);
        launch_gemm(cat_eh, attn_w_bf, attn_b, aw, nullptr, Bsz, Ls, H2, 0, stream);
        softmax32_kernel<<<Bsz, 32, 0, stream>>>(aw);
        attn_apply_cat_kernel<<<BH_BLOCKS, 256, 0, stream>>>(aw, enc_outs, emb_bf, cat_comb);
        launch_gemm(cat_comb, comb_w_bf, comb_b, nullptr, x_bf, Bsz, Hs, H2, 1, stream);
        launch_gemm(x_bf, dec_w_ih_bf, dec_b_ih, gi, nullptr, Bsz, H3, Hs, 0, stream);
        launch_gemm(h_bf, dec_w_hh_bf, dec_b_hh, gh, nullptr, Bsz, H3, Hs, 0, stream);
        gru_fuse_kernel<<<BH_BLOCKS, 256, 0, stream>>>(gi, gh, h_f32, h_bf, nullptr);
        launch_gemm(h_bf, out_w_bf, out_b, logits, nullptr, Bsz, Vs, Hs, 0, stream);
        log_softmax_kernel<<<Bsz, 256, 0, stream>>>(logits, out + (size_t)t * Bsz * Vs, Vs);
    }
}
